// Interval_Refine_75788992905508
// MI455X (gfx1250) — matrix operand order repaired and checked by reading, not compiled, not run on hardware
//
#include <hip/hip_runtime.h>
#include <hip/hip_bf16.h>
#include <math.h>

typedef __attribute__((ext_vector_type(16))) __bf16 v16bf;
typedef __attribute__((ext_vector_type(8)))  float  v8f;
typedef unsigned int u32x4 __attribute__((ext_vector_type(4)));
typedef int          i32x4 __attribute__((ext_vector_type(4)));
typedef int          i32x8 __attribute__((ext_vector_type(8)));

#define N_NODES 1024
#define D 128
#define THREE_D 384
#define DH 256
#define OUTC 69
#define SF 131

#ifndef __has_builtin
#define __has_builtin(x) 0
#endif
#if defined(__gfx1250__) && __has_builtin(__builtin_amdgcn_tensor_load_to_lds)
#define HAVE_TDM 1
#else
#define HAVE_TDM 0
#endif

__device__ __forceinline__ float sigmoidf_(float x) { return 1.0f / (1.0f + __expf(-x)); }

__device__ __forceinline__ v8f wmma_bf16(v16bf a, v16bf b, v8f c) {
  return __builtin_amdgcn_wmma_f32_16x16x32_bf16(false, a, false, b, (short)0, c, false, false);
}

// A fragment (16x32 bf16) from row-major bf16 [rows][row_stride], rows m0..m0+15, K chunk kc (32 wide).
__device__ __forceinline__ v16bf load_a_bf16(const __bf16* base, int row_stride, int m0, int kc) {
  int lane = threadIdx.x & 31;
  int half = lane >> 4;
  const __bf16* p = base + (size_t)(m0 + (lane & 15)) * row_stride + kc * 32;
  v16bf a;
#pragma unroll
  for (int e = 0; e < 8; ++e) a[e] = p[half * 8 + e];
#pragma unroll
  for (int e = 0; e < 8; ++e) a[8 + e] = p[16 + half * 8 + e];
  return a;
}

// Same but converting from f32 source on the fly.
__device__ __forceinline__ v16bf load_a_f32(const float* base, int row_stride, int m0, int kc) {
  int lane = threadIdx.x & 31;
  int half = lane >> 4;
  const float* p = base + (size_t)(m0 + (lane & 15)) * row_stride + kc * 32;
  v16bf a;
#pragma unroll
  for (int e = 0; e < 8; ++e) a[e] = (__bf16)p[half * 8 + e];
#pragma unroll
  for (int e = 0; e < 8; ++e) a[8 + e] = (__bf16)p[16 + half * 8 + e];
  return a;
}

// B fragment (32x16) for gh[m,j] = sum_k h[m,k]*W[j,k]: B[k][n] = W[n0+n][kc*32+k], W row-major [J][K].
__device__ __forceinline__ v16bf load_bT_bf16(const __bf16* w, int k_stride, int n0, int kc) {
  int lane = threadIdx.x & 31;
  const __bf16* p = w + (size_t)(n0 + (lane & 15)) * k_stride + kc * 32 + (lane >> 4) * 8;
  v16bf b;
#pragma unroll
  for (int e = 0; e < 8; ++e) { b[e] = p[e]; b[8 + e] = p[16 + e]; }
  return b;
}

__device__ __forceinline__ v16bf load_bT_f32(const float* w, int k_stride, int n0, int kc) {
  int lane = threadIdx.x & 31;
  const float* p = w + (size_t)(n0 + (lane & 15)) * k_stride + kc * 32 + (lane >> 4) * 8;
  v16bf b;
#pragma unroll
  for (int e = 0; e < 8; ++e) { b[e] = (__bf16)p[e]; b[8 + e] = (__bf16)p[16 + e]; }
  return b;
}

#if HAVE_TDM
// TDM: async DMA of one row of `nelem` f32 from global -> LDS (D# per CDNA5 ISA).
__device__ __forceinline__ void tdm_load_row_f32(unsigned lds_off, const float* gaddr, int nelem) {
  unsigned long long ga = (unsigned long long)(uintptr_t)gaddr;
  u32x4 g0;
  g0.x = 1u;                                                // count=1, user mode, no gather
  g0.y = lds_off;                                           // LDS byte address
  g0.z = (unsigned)(ga & 0xffffffffull);                    // global_addr[31:0]
  g0.w = (unsigned)((ga >> 32) & 0x01ffffffull) | 0x80000000u; // global_addr[56:32] | type=2
  i32x8 g1;
  g1[0] = (2 << 16);                                        // wg_mask=0, data_size=4B, no flags
  g1[1] = (nelem & 0xffff) << 16;                           // tensor_dim0 lo16 (atomic bar addr=0)
  g1[2] = ((nelem >> 16) & 0xffff) | (1 << 16);             // tensor_dim0 hi16 | tensor_dim1=1
  g1[3] = (nelem & 0xffff) << 16;                           // tensor_dim1 hi=0 | tile_dim0
  g1[4] = 1;                                                // tile_dim1=1, tile_dim2=0
  g1[5] = nelem;                                            // tensor_dim0_stride lo32
  g1[6] = (nelem & 0xffff) << 16;                           // stride0 hi16=0 | stride1 lo16
  g1[7] = 0;                                                // stride1 hi
  i32x4 z4 = {0, 0, 0, 0};
#if __clang_major__ >= 23
  i32x8 z8 = {0, 0, 0, 0, 0, 0, 0, 0};
  __builtin_amdgcn_tensor_load_to_lds(g0, g1, z4, z4, z8, 0);
#else
  __builtin_amdgcn_tensor_load_to_lds(g0, g1, z4, z4, 0);
#endif
}
__device__ __forceinline__ void tdm_wait() {
#if __has_builtin(__builtin_amdgcn_s_wait_tensorcnt)
  __builtin_amdgcn_s_wait_tensorcnt(0);
#endif
}
#endif

// ---------- kernel 1: h_relu = relu(E @ W_adapt.T + b_adapt), WMMA ----------
__global__ void __launch_bounds__(256) k_adapt(const float* __restrict__ E,
                                               const float* __restrict__ W,
                                               const float* __restrict__ bias,
                                               float* __restrict__ hrelu) {
  int mi = blockIdx.x;            // 64 row strips of 16
  int wv = threadIdx.x >> 5;      // wave = N-tile (8 tiles of 16 cover D=128)
  int lane = threadIdx.x & 31;
  v8f acc = {};
#pragma unroll
  for (int kc = 0; kc < 4; ++kc) {
    v16bf a = load_a_f32(E, D, mi * 16, kc);
    v16bf b = load_bT_f32(W, D, wv * 16, kc);
    acc = wmma_bf16(a, b, acc);
  }
  int half = lane >> 4, lo = lane & 15;
  int col = wv * 16 + lo;
  float bc = bias[col];
#pragma unroll
  for (int r = 0; r < 8; ++r) {
    int row = mi * 16 + half * 8 + r;
    float v = acc[r] + bc;
    hrelu[(size_t)row * D + col] = v > 0.f ? v : 0.f;
  }
}

// ---------- kernel 2: LayerNorm + sinusoid -> x (bf16) ----------
__global__ void __launch_bounds__(128) k_ln(const float* __restrict__ hrelu,
                                            const float* __restrict__ tpos,
                                            const float* __restrict__ g,
                                            const float* __restrict__ bta,
                                            __bf16* __restrict__ xbf) {
  __shared__ float s1[128], s2[128];
  int r = blockIdx.x, k = threadIdx.x;
  float v = hrelu[(size_t)r * D + k];
  s1[k] = v; s2[k] = v * v;
  __syncthreads();
  for (int off = 64; off > 0; off >>= 1) {
    if (k < off) { s1[k] += s1[k + off]; s2[k] += s2[k + off]; }
    __syncthreads();
  }
  float mu = s1[0] * (1.0f / 128.0f);
  float var = s2[0] * (1.0f / 128.0f) - mu * mu;
  float nea = (v - mu) * rsqrtf(var + 1e-5f) * g[k] + bta[k];
  float freq = 10.0f * (float)k / 127.0f;
  float x = nea + 0.05f * __sinf(tpos[r] * freq);
  xbf[(size_t)r * D + k] = (__bf16)x;
}

// ---------- kernel 3: smoothing + softmax -> abnormal ----------
__global__ void k_smooth(const float* __restrict__ np_,
                         const float* __restrict__ sw,
                         float* __restrict__ abn) {
  int i = blockIdx.x * blockDim.x + threadIdx.x;
  if (i >= N_NODES) return;
  float l[5];
#pragma unroll
  for (int c = 0; c < 5; ++c) {
    float s = 0.f;
#pragma unroll
    for (int j = 0; j < 5; ++j) {
      int idx = i - 2 + j;
      if (idx >= 0 && idx < N_NODES) s += np_[idx * 5 + c] * sw[c * 5 + j];
    }
    l[c] = s;
  }
  float mx = l[0];
#pragma unroll
  for (int c = 1; c < 5; ++c) mx = fmaxf(mx, l[c]);
  float sum = 0.f, e0 = 0.f;
#pragma unroll
  for (int c = 0; c < 5; ++c) { float e = __expf(l[c] - mx); sum += e; if (c == 0) e0 = e; }
  abn[i] = e0 / sum;
}

// ---------- kernel 4: f32 -> bf16 convert (whh) ----------
__global__ void k_cvt(const float* __restrict__ src, __bf16* __restrict__ dst, int n) {
  int i = blockIdx.x * blockDim.x + threadIdx.x;
  if (i < n) dst[i] = (__bf16)src[i];
}

// ---------- kernel 5: gi[s] = x @ wih[s].T + bih[s], WMMA ----------
__global__ void __launch_bounds__(256) k_gi(const __bf16* __restrict__ xbf,
                                            const float* __restrict__ wih,
                                            const float* __restrict__ bih,
                                            float* __restrict__ gi) {
  int s = blockIdx.x / 64, mi = blockIdx.x % 64;
  int wv = threadIdx.x >> 5, lane = threadIdx.x & 31;
  const float* w = wih + (size_t)s * THREE_D * D;
  const float* bi = bih + (size_t)s * THREE_D;
  float* go = gi + (size_t)s * N_NODES * THREE_D;
  for (int nj = wv; nj < 24; nj += 8) {
    v8f acc = {};
#pragma unroll
    for (int kc = 0; kc < 4; ++kc) {
      v16bf a = load_a_bf16(xbf, D, mi * 16, kc);
      v16bf b = load_bT_f32(w, D, nj * 16, kc);
      acc = wmma_bf16(a, b, acc);
    }
    int half = lane >> 4, lo = lane & 15;
    int col = nj * 16 + lo;
    float bc = bi[col];
#pragma unroll
    for (int r = 0; r < 8; ++r) {
      int row = mi * 16 + half * 8 + r;
      go[(size_t)row * THREE_D + col] = acc[r] + bc;
    }
  }
}

// ---------- kernel 6: scalar abnormal GRU (H=1), one thread per anchor ----------
__global__ void k_agru(const float* __restrict__ abn, const float* __restrict__ anchors,
                       const float* __restrict__ tp, const float* __restrict__ alen_p,
                       const float* __restrict__ wih, const float* __restrict__ whh,
                       const float* __restrict__ bih, const float* __restrict__ bhh,
                       float* __restrict__ ha) {
  int a = threadIdx.x;
  if (a >= 96) return;
  int s = a < 48 ? 0 : (a < 80 ? 1 : 2);
  float alen = alen_p[0];
  float st = anchors[a * 2], en = anchors[a * 2 + 1];
  const float* wi = wih + s * 3;
  const float* wh = whh + s * 3;
  const float* bi = bih + s * 3;
  const float* bh = bhh + s * 3;
  float h = 0.f; bool any = false;
  for (int t = 0; t < N_NODES; ++t) {
    float tpos = tp[t] * alen;
    if (tpos >= st && tpos <= en) {
      float x = abn[t];
      float r = sigmoidf_(x * wi[0] + bi[0] + h * wh[0] + bh[0]);
      float z = sigmoidf_(x * wi[1] + bi[1] + h * wh[1] + bh[1]);
      float n = tanhf(x * wi[2] + bi[2] + r * (h * wh[2] + bh[2]));
      h = (1.f - z) * n + z * h;
      any = true;
    }
  }
  if (!any) {
    float r0 = sigmoidf_(bi[0] + bh[0]);
    float z0 = sigmoidf_(bi[1] + bh[1]);
    h = (1.f - z0) * tanhf(bi[2] + r0 * bh[2]);
  }
  ha[a] = h;
}

// ---------- kernel 7: persistent masked GRU scan, one workgroup per scale ----------
// Per step: WMMA gh = h @ whh.T; TDM prefetches next gi row into an LDS double buffer,
// overlapped with matrix work and synchronized with s_wait_tensorcnt + workgroup barrier.
__global__ void __launch_bounds__(256) k_gru_scan(const __bf16* __restrict__ whh_bf,
                                                  const float* __restrict__ gi,
                                                  const float* __restrict__ bhh_all,
                                                  const float* __restrict__ bih_all,
                                                  const float* __restrict__ anchors,
                                                  const float* __restrict__ tp,
                                                  const float* __restrict__ alen_p,
                                                  float* __restrict__ hfinal) {
  __shared__ __bf16 h_bf[48 * 128];     // hidden state (bf16 copy for WMMA A)
  __shared__ __bf16 gh_bf[48 * 384];    // per-step gate pre-activations (h @ whh.T)
  __shared__ float sstart[48], send[48];
  __shared__ float gibuf[2][THREE_D];   // TDM double buffer for gi rows
  int s = blockIdx.x;
  int M = (s == 0) ? 48 : ((s == 1) ? 32 : 16);
  int aoff = (s == 0) ? 0 : ((s == 1) ? 48 : 80);
  int tid = threadIdx.x;
  float alen = alen_p[0];
  const __bf16* whh = whh_bf + (size_t)s * THREE_D * D;
  const float* gis = gi + (size_t)s * N_NODES * THREE_D;
  const float* bhh = bhh_all + s * THREE_D;
  const float* bih = bih_all + s * THREE_D;

  for (int i = tid; i < M * D; i += 256) h_bf[i] = (__bf16)0.0f;
  if (tid < M) { sstart[tid] = anchors[(aoff + tid) * 2]; send[tid] = anchors[(aoff + tid) * 2 + 1]; }
  int nown = (M * D) / 256;             // 24 / 16 / 8 elements owned per thread
  float hreg[24];
  bool anym[24];
  for (int i = 0; i < 24; ++i) { hreg[i] = 0.f; anym[i] = false; }

#if HAVE_TDM
  if (tid < 32) {                       // wave 0 drives the TDM pipeline
    tdm_load_row_f32((unsigned)(uintptr_t)&gibuf[0][0], gis, THREE_D);
  }
#endif
  __syncthreads();

  int wv = tid >> 5;
  int lane = tid & 31;
  int half = lane >> 4, lo = lane & 15;
  int ntiles = (M / 16) * 24;           // (M/16) M-tiles x 24 N-tiles over 384 cols

  for (int t = 0; t < N_NODES; ++t) {
    // phase 1: gh = h @ whh.T via WMMA (uniform control flow per wave; EXEC all-1s)
    for (int tile = wv; tile < ntiles; tile += 8) {
      int mi = tile / 24, nj = tile % 24;
      if (tile + 8 < ntiles)            // hint next B tile into cache
        __builtin_prefetch(whh + (size_t)(((tile + 8) % 24) * 16) * D, 0, 0);
      v8f acc = {};
#pragma unroll
      for (int kc = 0; kc < 4; ++kc) {
        v16bf a = load_a_bf16(h_bf, D, mi * 16, kc);
        v16bf b = load_bT_bf16(whh, D, nj * 16, kc);
        acc = wmma_bf16(a, b, acc);
      }
#pragma unroll
      for (int r = 0; r < 8; ++r) {
        int m = mi * 16 + half * 8 + r;
        gh_bf[m * THREE_D + nj * 16 + lo] = (__bf16)acc[r];
      }
    }
#if HAVE_TDM
    if (tid < 32) tdm_wait();           // gi[t] now resident in gibuf[t&1]
#endif
    __syncthreads();
#if HAVE_TDM
    if (tid < 32 && (t + 1) < N_NODES) {  // prefetch gi[t+1] into the other buffer
      tdm_load_row_f32((unsigned)(uintptr_t)&gibuf[(t + 1) & 1][0],
                       gis + (size_t)(t + 1) * THREE_D, THREE_D);
    }
    const float* girow = &gibuf[t & 1][0];
#else
    const float* girow = gis + (size_t)t * THREE_D;
#endif
    // phase 2: gates + masked state update (equivalent to where(mask, h_new, h))
    float tpos = tp[t] * alen;
    for (int i = 0; i < nown; ++i) {
      int idx = tid + i * 256;
      int m = idx >> 7, k = idx & 127;
      if (tpos >= sstart[m] && tpos <= send[m]) {
        float ghr = (float)gh_bf[m * THREE_D + k] + bhh[k];
        float ghz = (float)gh_bf[m * THREE_D + 128 + k] + bhh[128 + k];
        float ghg = (float)gh_bf[m * THREE_D + 256 + k] + bhh[256 + k];
        float r = sigmoidf_(girow[k] + ghr);
        float z = sigmoidf_(girow[128 + k] + ghz);
        float n = tanhf(girow[256 + k] + r * ghg);
        float hn = (1.f - z) * n + z * hreg[i];
        hreg[i] = hn;
        h_bf[idx] = (__bf16)hn;
        anym[i] = true;
      }
    }
    __syncthreads();
  }
  for (int i = 0; i < nown; ++i) {
    int idx = tid + i * 256;
    int m = idx >> 7, k = idx & 127;
    float h = hreg[i];
    if (!anym[i]) {   // empty-interval anchor: reference's h_empty
      float r0 = sigmoidf_(bih[k] + bhh[k]);
      float z0 = sigmoidf_(bih[128 + k] + bhh[128 + k]);
      h = (1.f - z0) * tanhf(bih[256 + k] + r0 * bhh[256 + k]);
    }
    hfinal[(size_t)(aoff + m) * D + k] = h;
  }
}

// ---------- kernel 8: head MLP + bin softmax + output assembly ----------
__global__ void __launch_bounds__(256) k_head(const float* __restrict__ hfinal,
                                              const float* __restrict__ ha,
                                              const float* __restrict__ anchors,
                                              const float* __restrict__ alen_p,
                                              const float* __restrict__ w1, const float* __restrict__ b1,
                                              const float* __restrict__ w2, const float* __restrict__ b2,
                                              const float* __restrict__ w3, const float* __restrict__ b3,
                                              const float* __restrict__ wparams,
                                              float* __restrict__ out) {
  __shared__ float sf[SF];
  __shared__ float f1[DH];
  __shared__ float f2[DH];
  __shared__ float lg[OUTC];
  int a = blockIdx.x;
  int s = a < 48 ? 0 : (a < 80 ? 1 : 2);
  int tid = threadIdx.x;
  float alen = alen_p[0];
  float st = anchors[a * 2], en = anchors[a * 2 + 1];
  if (tid < 128) sf[tid] = hfinal[(size_t)a * D + tid];
  if (tid == 128) sf[128] = ha[a];
  if (tid == 129) sf[129] = ((st + en) * 0.5f) / alen;
  if (tid == 130) sf[130] = (en - st) / alen;
  __syncthreads();
  {
    const float* row = w1 + ((size_t)s * DH + tid) * SF;
    float acc = b1[s * DH + tid];
    for (int k = 0; k < SF; ++k) acc += row[k] * sf[k];
    f1[tid] = fmaxf(acc, 0.f);
  }
  __syncthreads();
  {
    const float* row = w2 + ((size_t)s * DH + tid) * DH;
    float acc = b2[s * DH + tid];
    for (int k = 0; k < DH; ++k) acc += row[k] * f1[k];
    f2[tid] = fmaxf(acc, 0.f);
  }
  __syncthreads();
  if (tid < OUTC) {
    const float* row = w3 + ((size_t)s * OUTC + tid) * DH;
    float acc = b3[s * OUTC + tid];
    for (int k = 0; k < DH; ++k) acc += row[k] * f2[k];
    lg[tid] = acc;
  }
  __syncthreads();
  if (tid == 0) {
    const float* wp = wparams + s * 32;
    float so = 0.f, eo = 0.f;
    for (int g = 0; g < 2; ++g) {
      const float* L = lg + g * 32;
      float mx = L[0];
      for (int i = 1; i < 32; ++i) mx = fmaxf(mx, L[i]);
      float sum = 0.f, ws = 0.f;
      for (int i = 0; i < 32; ++i) { float e = __expf(L[i] - mx); sum += e; ws += e * wp[i]; }
      float o = ws / sum;
      if (g == 0) so = o; else eo = o;
    }
    float stc = fminf(fmaxf(st + so, 0.f), alen);
    float enc = fminf(fmaxf(en + eo, 0.f), alen);
    float* o = out + a * 7;
    o[0] = stc; o[1] = enc; o[2] = lg[64];
    o[3] = lg[65]; o[4] = lg[66]; o[5] = lg[67]; o[6] = lg[68];
  }
}

extern "C" void kernel_launch(void* const* d_in, const int* in_sizes, int n_in,
                              void* d_out, int out_size, void* d_ws, size_t ws_size,
                              hipStream_t stream) {
  (void)in_sizes; (void)n_in; (void)out_size; (void)ws_size;
  const float* E    = (const float*)d_in[0];
  const float* tp   = (const float*)d_in[1];
  const float* np_  = (const float*)d_in[2];
  const float* alen = (const float*)d_in[3];
  const float* anch = (const float*)d_in[4];
  const float* sw   = (const float*)d_in[5];
  const float* Wad  = (const float*)d_in[6];
  const float* bad  = (const float*)d_in[7];
  const float* lng  = (const float*)d_in[8];
  const float* lnb  = (const float*)d_in[9];
  const float* gwih = (const float*)d_in[10];
  const float* gwhh = (const float*)d_in[11];
  const float* gbih = (const float*)d_in[12];
  const float* gbhh = (const float*)d_in[13];
  const float* awih = (const float*)d_in[14];
  const float* awhh = (const float*)d_in[15];
  const float* abih = (const float*)d_in[16];
  const float* abhh = (const float*)d_in[17];
  const float* wpar = (const float*)d_in[18];
  const float* w1   = (const float*)d_in[19];
  const float* b1   = (const float*)d_in[20];
  const float* w2   = (const float*)d_in[21];
  const float* b2   = (const float*)d_in[22];
  const float* w3   = (const float*)d_in[23];
  const float* b3   = (const float*)d_in[24];
  float* out = (float*)d_out;

  char* ws = (char*)d_ws;
  float*  hrelu = (float*)(ws + 0);                  // 1024*128*4   = 524288
  __bf16* xbf   = (__bf16*)(ws + 524288);            // 1024*128*2   = 262144
  float*  gi    = (float*)(ws + 786432);             // 3*1024*384*4 = 4718592
  __bf16* whhbf = (__bf16*)(ws + 5505024);           // 3*384*128*2  = 294912
  float*  abn   = (float*)(ws + 5799936);            // 1024*4       = 4096
  float*  hfin  = (float*)(ws + 5804032);            // 96*128*4     = 49152
  float*  ha    = (float*)(ws + 5853184);            // 96*4         = 384

  hipLaunchKernelGGL(k_adapt,   dim3(64),   dim3(256), 0, stream, E, Wad, bad, hrelu);
  hipLaunchKernelGGL(k_ln,      dim3(1024), dim3(128), 0, stream, hrelu, tp, lng, lnb, xbf);
  hipLaunchKernelGGL(k_smooth,  dim3(4),    dim3(256), 0, stream, np_, sw, abn);
  hipLaunchKernelGGL(k_cvt,     dim3(576),  dim3(256), 0, stream, gwhh, whhbf, 3 * THREE_D * D);
  hipLaunchKernelGGL(k_gi,      dim3(192),  dim3(256), 0, stream, xbf, gwih, gbih, gi);
  hipLaunchKernelGGL(k_agru,    dim3(1),    dim3(96),  0, stream, abn, anch, tp, alen, awih, awhh, abih, abhh, ha);
  hipLaunchKernelGGL(k_gru_scan,dim3(3),    dim3(256), 0, stream, whhbf, gi, gbhh, gbih, anch, tp, alen, hfin);
  hipLaunchKernelGGL(k_head,    dim3(96),   dim3(256), 0, stream, hfin, ha, anch, alen, w1, b1, w2, b2, w3, b3, wpar, out);
}
